// VGAE_8186207666838
// MI455X (gfx1250) — matrix operand order repaired and checked by reading, not compiled, not run on hardware
//
#include <hip/hip_runtime.h>

#define N_NODES 10000
#define N_EDGES 320000
#define IN_F    512
#define OUT_F   64
#define NT      625   // N_NODES / 16 (exact)

typedef __attribute__((ext_vector_type(16))) _Float16 v16h;
typedef __attribute__((ext_vector_type(8)))  _Float16 v8h;
typedef __attribute__((ext_vector_type(8)))  float    v8f;

// ---------------- degree computation ----------------
__global__ void vgae_degree(const int* __restrict__ src, const int* __restrict__ dst,
                            float* cnt_out, float* cnt_in) {
  int e = blockIdx.x * blockDim.x + threadIdx.x;
  if (e < N_EDGES) {
    atomicAdd(&cnt_out[src[e]], 1.0f);
    atomicAdd(&cnt_in [dst[e]], 1.0f);
  }
}

__global__ void vgae_finalize_deg(float* s_out, float* s_in) {
  int i = blockIdx.x * blockDim.x + threadIdx.x;
  if (i < N_NODES) {
    s_out[i] = rsqrtf(fmaxf(s_out[i], 1.0f));
    s_in [i] = rsqrtf(fmaxf(s_in [i], 1.0f));
  }
}

// ---------------- layer 1: (feat * D_out^-1/2) @ W1 ----------------
__global__ void vgae_gemm1(const float* __restrict__ feat, const float* __restrict__ W1,
                           const float* __restrict__ s_out, float* __restrict__ hproj) {
  int gid  = blockIdx.x * blockDim.x + threadIdx.x;
  int node = gid >> 6;
  int o    = gid & 63;
  if (node >= N_NODES) return;
  const float* fr = feat + (size_t)node * IN_F;
  float acc = 0.0f;
#pragma unroll 8
  for (int k = 0; k < IN_F; ++k) acc += fr[k] * W1[k * OUT_F + o];
  hproj[gid] = acc * s_out[node];
}

// ---------------- edge aggregation (segment_sum of h[src] into dst) ----------------
__global__ void vgae_scatter1(const int* __restrict__ src, const int* __restrict__ dst,
                              const float* __restrict__ hproj, float* __restrict__ agg) {
  int gid = blockIdx.x * blockDim.x + threadIdx.x;   // < N_EDGES*64
  int e = gid >> 6;
  int o = gid & 63;
  atomicAdd(&agg[(size_t)dst[e] * OUT_F + o], hproj[(size_t)src[e] * OUT_F + o]);
}

__global__ void vgae_finalize1(const float* __restrict__ agg, const float* __restrict__ s_in,
                               const float* __restrict__ b1, float* __restrict__ h1) {
  int gid  = blockIdx.x * blockDim.x + threadIdx.x;
  int node = gid >> 6;
  int o    = gid & 63;
  if (node >= N_NODES) return;
  h1[gid] = agg[gid] * s_in[node] + b1[o];
}

// ---------------- layers 2/3 projections (fused) ----------------
__global__ void vgae_gemm23(const float* __restrict__ h1,
                            const float* __restrict__ Wmu, const float* __restrict__ Wstd,
                            const float* __restrict__ s_out,
                            float* __restrict__ muproj, float* __restrict__ stdproj) {
  int gid  = blockIdx.x * blockDim.x + threadIdx.x;
  int node = gid >> 6;
  int o    = gid & 63;
  if (node >= N_NODES) return;
  const float* hr = h1 + (size_t)node * OUT_F;
  float am = 0.0f, as = 0.0f;
#pragma unroll 8
  for (int k = 0; k < OUT_F; ++k) {
    float hv = hr[k];
    am += hv * Wmu [k * OUT_F + o];
    as += hv * Wstd[k * OUT_F + o];
  }
  float s = s_out[node];
  muproj [gid] = am * s;
  stdproj[gid] = as * s;
}

__global__ void vgae_scatter23(const int* __restrict__ src, const int* __restrict__ dst,
                               const float* __restrict__ muproj, const float* __restrict__ stdproj,
                               float* __restrict__ aggmu, float* __restrict__ aggstd) {
  int gid = blockIdx.x * blockDim.x + threadIdx.x;   // < N_EDGES*64
  int e = gid >> 6;
  int o = gid & 63;
  size_t di = (size_t)dst[e] * OUT_F + o;
  size_t si = (size_t)src[e] * OUT_F + o;
  atomicAdd(&aggmu [di], muproj [si]);
  atomicAdd(&aggstd[di], stdproj[si]);
}

// ---------------- reparametrize: mu, std=exp(logvar), z = eps*std + mu ----------------
__global__ void vgae_finalize2(const float* __restrict__ aggmu, const float* __restrict__ aggstd,
                               const float* __restrict__ s_in,
                               const float* __restrict__ bmu, const float* __restrict__ bstd,
                               const float* __restrict__ eps,
                               float* __restrict__ out_mu, float* __restrict__ out_std,
                               _Float16* __restrict__ zh) {
  int gid  = blockIdx.x * blockDim.x + threadIdx.x;
  int node = gid >> 6;
  int o    = gid & 63;
  if (node >= N_NODES) return;
  float si = s_in[node];
  float mu = aggmu [gid] * si + bmu [o];
  float lv = aggstd[gid] * si + bstd[o];
  float sd = expf(lv);
  float z  = eps[gid] * sd + mu;
  out_mu [gid] = mu;
  out_std[gid] = sd;
  zh[gid] = (_Float16)z;
}

// ---------------- dense decode: adj = sigmoid(Z @ Z^T) via WMMA f16 ----------------
// One wave32 per 16x16 output tile; K = 64 -> two v_wmma_f32_16x16x32_f16.
// A (16x32 f16) lane layout: M = lane&15, kbase = (lane>>4)*8, VGPR0-3 = K+0..7, VGPR4-7 = K+16..23.
// B (32x16 f16) lane layout: N = lane&15, kbase = (lane>>4)*8, VGPR0-3 = K+0..7, VGPR4-7 = K+16..23.
// Since C = Z*Z^T, B[:,n] is row n of Z -> both operands are contiguous 16B loads from zh.
// Sigmoid uses the hardware rcp (v_exp + v_add + v_rcp) instead of the IEEE
// division sequence: 1e8 outputs make the divide chain the VALU bottleneck.
__global__ __launch_bounds__(256) void vgae_decode(const _Float16* __restrict__ zh,
                                                   float* __restrict__ adj) {
  int lane = threadIdx.x & 31;
  int wave = threadIdx.x >> 5;
  int jt = blockIdx.x * 8 + wave;   // column tile
  int it = blockIdx.y;              // row tile
  if (jt >= NT) return;             // uniform per wave -> active waves keep EXEC all-1s

  int half = lane >> 4;             // 0 or 1
  int l15  = lane & 15;
  const _Float16* za = zh + (size_t)(it * 16 + l15) * OUT_F;  // A: row it*16+M
  const _Float16* zb = zh + (size_t)(jt * 16 + l15) * OUT_F;  // B: column jt*16+N == row of Z
  int kbA = half * 8;
  int kbB = half * 8;

  v8f c = {};
#pragma unroll
  for (int kc = 0; kc < OUT_F; kc += 32) {
    v8h alo = *(const v8h*)(za + kc + kbA);        // K + 0..7
    v8h ahi = *(const v8h*)(za + kc + kbA + 16);   // K + 16..23
    v8h blo = *(const v8h*)(zb + kc + kbB);        // K + 0..7
    v8h bhi = *(const v8h*)(zb + kc + kbB + 16);   // K + 16..23
    v16h a, b;
#pragma unroll
    for (int i = 0; i < 8; ++i) {
      a[i] = alo[i]; a[i + 8] = ahi[i];
      b[i] = blo[i]; b[i + 8] = bhi[i];
    }
    c = __builtin_amdgcn_wmma_f32_16x16x32_f16(
            /*neg_a=*/false, a, /*neg_b=*/false, b,
            /*c_mod=*/(short)0, c, /*reuse_a=*/false, /*reuse_b=*/false);
  }

  int col = jt * 16 + l15;          // N = lane&15
#pragma unroll
  for (int r = 0; r < 8; ++r) {
    int row = it * 16 + r + half * 8;   // M = r + 8*(lane>>4)
    // fast sigmoid: v_exp_f32 + v_add_f32 + v_rcp_f32 (no IEEE div sequence)
    float e = __expf(-c[r]);
    adj[(size_t)row * N_NODES + col] = __builtin_amdgcn_rcpf(1.0f + e);
  }
}

extern "C" void kernel_launch(void* const* d_in, const int* in_sizes, int n_in,
                              void* d_out, int out_size, void* d_ws, size_t ws_size,
                              hipStream_t stream) {
  (void)in_sizes; (void)n_in; (void)out_size; (void)ws_size;

  const float* feat = (const float*)d_in[0];
  const float* eps  = (const float*)d_in[1];
  const int*   src  = (const int*)  d_in[2];
  const int*   dst  = (const int*)  d_in[3];
  const float* W1   = (const float*)d_in[4];
  const float* b1   = (const float*)d_in[5];
  const float* Wmu  = (const float*)d_in[6];
  const float* bmu  = (const float*)d_in[7];
  const float* Wstd = (const float*)d_in[8];
  const float* bstd = (const float*)d_in[9];

  float* out     = (float*)d_out;
  float* adj     = out;                                    // [N, N]
  float* out_mu  = out + (size_t)N_NODES * N_NODES;        // [N, 64]
  float* out_std = out_mu + (size_t)N_NODES * OUT_F;       // [N, 64]

  // Scratch lives in the 400MB adj region: everything here is dead before
  // vgae_decode overwrites adj. Only zh (read by decode) lives in d_ws.
  float* s_out   = adj;                                    // N       (zeroed)
  float* s_in    = s_out   + N_NODES;                      // N       (zeroed)
  float* agg1    = s_in    + N_NODES;                      // 64N     (zeroed)
  float* aggmu   = agg1    + (size_t)N_NODES * OUT_F;      // 64N     (zeroed)
  float* aggstd  = aggmu   + (size_t)N_NODES * OUT_F;      // 64N     (zeroed)
  float* hproj   = aggstd  + (size_t)N_NODES * OUT_F;      // 64N
  float* h1      = hproj   + (size_t)N_NODES * OUT_F;      // 64N
  float* muproj  = h1      + (size_t)N_NODES * OUT_F;      // 64N
  float* stdproj = muproj  + (size_t)N_NODES * OUT_F;      // 64N
  _Float16* zh   = (_Float16*)d_ws;                        // N*64 f16 = 1.28MB

  // zero the accumulator region (s_out..aggstd are contiguous)
  size_t zero_bytes = (size_t)(2 * N_NODES + 3 * N_NODES * OUT_F) * sizeof(float);
  hipMemsetAsync(s_out, 0, zero_bytes, stream);

  const int TPB = 256;
  vgae_degree      <<<(N_EDGES + TPB - 1) / TPB, TPB, 0, stream>>>(src, dst, s_out, s_in);
  vgae_finalize_deg<<<(N_NODES + TPB - 1) / TPB, TPB, 0, stream>>>(s_out, s_in);

  int nElems = N_NODES * OUT_F;                 // 640000, divisible by 256
  vgae_gemm1    <<<nElems / TPB, TPB, 0, stream>>>(feat, W1, s_out, hproj);

  int nEdgeWork = N_EDGES * OUT_F;              // 20,480,000, divisible by 256
  vgae_scatter1 <<<nEdgeWork / TPB, TPB, 0, stream>>>(src, dst, hproj, agg1);
  vgae_finalize1<<<nElems / TPB, TPB, 0, stream>>>(agg1, s_in, b1, h1);

  vgae_gemm23   <<<nElems / TPB, TPB, 0, stream>>>(h1, Wmu, Wstd, s_out, muproj, stdproj);
  vgae_scatter23<<<nEdgeWork / TPB, TPB, 0, stream>>>(src, dst, muproj, stdproj, aggmu, aggstd);
  vgae_finalize2<<<nElems / TPB, TPB, 0, stream>>>(aggmu, aggstd, s_in, bmu, bstd, eps,
                                                   out_mu, out_std, zh);

  // decode: 625x625 tiles of 16x16, 8 waves (tiles) per block
  dim3 grid((NT + 7) / 8, NT);
  vgae_decode<<<grid, TPB, 0, stream>>>(zh, adj);
}
